// RWKV6Attention_34540126994590
// MI455X (gfx1250) — matrix operand order repaired and checked by reading, not compiled, not run on hardware
//
#include <hip/hip_runtime.h>

// ---------------------------------------------------------------------------
// RWKV6 forward for MI455X (gfx1250). B=2,T=1024,C=1024,H=16,K=64,A=32,D=64.
// Big GEMMs: v_wmma_f32_16x16x32_bf16, bf16x3 fp32 emulation, tiles staged
// with GLOBAL_LOAD_ASYNC_TO_LDS_B128 (double-buffered, ASYNCcnt-synced).
// ---------------------------------------------------------------------------

typedef __attribute__((ext_vector_type(16))) __bf16 v16bf;
typedef __attribute__((ext_vector_type(8)))  float  v8f;

union Frag { v16bf v; uint4 u4[2]; };

__device__ inline void bf16_split(float f, unsigned short &hi, unsigned short &lo) {
    unsigned u = __float_as_uint(f);
    unsigned short h = (unsigned short)(u >> 16);            // truncated hi plane
    float fh = __uint_as_float(((unsigned)h) << 16);
    float r = f - fh;                                        // residual
    unsigned ur = __float_as_uint(r);
    unsigned short l = (unsigned short)((ur + 0x7fffu + ((ur >> 16) & 1u)) >> 16); // RNE
    hi = h; lo = l;
}

// ---------------------------------------------------------------------------
// Kernel 1: token shift. sx = prev - x ; xxx = x + sx * time_maa_x
// ---------------------------------------------------------------------------
__global__ void k_mix1(const float* __restrict__ x, const float* __restrict__ tmx,
                       float* __restrict__ sx, float* __restrict__ xxx) {
    int idx = blockIdx.x * 256 + threadIdx.x;      // over M*C = 2048*1024
    int c = idx & 1023;
    int t = (idx >> 10) & 1023;                    // row = b*T + t, T = 1024
    float xv = x[idx];
    float prev = (t == 0) ? 0.f : x[idx - 1024];
    float s = prev - xv;
    sx[idx] = s;
    xxx[idx] = xv + s * tmx[c];
}

// ---------------------------------------------------------------------------
// Kernel 2: m = tanh(xxx @ tm_w1)  -> [M,160]
// ---------------------------------------------------------------------------
__global__ void k_lora1(const float* __restrict__ xxx, const float* __restrict__ w1,
                        float* __restrict__ mout) {
    int row = blockIdx.x;          // 0..2047
    int col = threadIdx.x;         // 0..159
    const float* xr = xxx + (size_t)row * 1024;
    float acc = 0.f;
    for (int k = 0; k < 1024; ++k) acc += xr[k] * w1[k * 160 + col];
    mout[row * 160 + col] = tanhf(acc);
}

// ---------------------------------------------------------------------------
// Kernel 3: delta = m @ tm_w2 ; xi = x + sx*(mu_i + delta_i), i in {w,k,v,r,g}
// ---------------------------------------------------------------------------
__global__ void k_mix2(const float* __restrict__ x, const float* __restrict__ sx,
                       const float* __restrict__ m, const float* __restrict__ mu5,
                       const float* __restrict__ w2,
                       float* __restrict__ xw, float* __restrict__ xk,
                       float* __restrict__ xv, float* __restrict__ xr,
                       float* __restrict__ xg) {
    __shared__ float sm[160];
    int row = blockIdx.y;
    int c = blockIdx.x * 256 + threadIdx.x;
    if (threadIdx.x < 160) sm[threadIdx.x] = m[row * 160 + threadIdx.x];
    __syncthreads();
    size_t idx = (size_t)row * 1024 + c;
    float xval = x[idx], s = sx[idx];
    float out[5];
#pragma unroll
    for (int f = 0; f < 5; ++f) {
        float d = 0.f;
        const float* wf = w2 + (size_t)(f * 32) * 1024 + c;
#pragma unroll
        for (int a = 0; a < 32; ++a) d += sm[f * 32 + a] * wf[(size_t)a * 1024];
        out[f] = xval + s * (mu5[f * 1024 + c] + d);
    }
    xw[idx] = out[0]; xk[idx] = out[1]; xv[idx] = out[2]; xr[idx] = out[3]; xg[idx] = out[4];
}

// ---------------------------------------------------------------------------
// Kernel 4: t1 = tanh(xw @ td_w1)   [M,64]
// ---------------------------------------------------------------------------
__global__ void k_decay1(const float* __restrict__ xw, const float* __restrict__ w1,
                         float* __restrict__ t1) {
    int row = blockIdx.x;
    int col = threadIdx.x;          // 0..63
    const float* xr = xw + (size_t)row * 1024;
    float acc = 0.f;
    for (int k = 0; k < 1024; ++k) acc += xr[k] * w1[k * 64 + col];
    t1[row * 64 + col] = tanhf(acc);
}

// ---------------------------------------------------------------------------
// Kernel 5: w = exp(-exp(time_decay + t1 @ td_w2))
// ---------------------------------------------------------------------------
__global__ void k_decay2(const float* __restrict__ t1, const float* __restrict__ w2,
                         const float* __restrict__ td, float* __restrict__ wout) {
    __shared__ float st[64];
    int row = blockIdx.y;
    int c = blockIdx.x * 256 + threadIdx.x;
    if (threadIdx.x < 64) st[threadIdx.x] = t1[row * 64 + threadIdx.x];
    __syncthreads();
    float acc = 0.f;
#pragma unroll
    for (int a = 0; a < 64; ++a) acc += st[a] * w2[(size_t)a * 1024 + c];
    float ww = td[c] + acc;
    wout[(size_t)row * 1024 + c] = expf(-expf(ww));
}

// ---------------------------------------------------------------------------
// Conversion: f32 activations [2048,1024] -> bf16 hi/lo planes (row-major)
// ---------------------------------------------------------------------------
__global__ void k_conv_act(const float* __restrict__ src,
                           unsigned short* __restrict__ hi,
                           unsigned short* __restrict__ lo) {
    int i2 = blockIdx.x * 256 + threadIdx.x;       // over MC/2, 2 elems each
    float2 f = *(const float2*)(src + (size_t)i2 * 2);
    unsigned short h0, l0, h1, l1;
    bf16_split(f.x, h0, l0);
    bf16_split(f.y, h1, l1);
    ((unsigned*)hi)[i2] = (unsigned)h0 | ((unsigned)h1 << 16);
    ((unsigned*)lo)[i2] = (unsigned)l0 | ((unsigned)l1 << 16);
}

// ---------------------------------------------------------------------------
// Conversion: f32 weight W[k][n] (1024x1024) -> TRANSPOSED bf16 planes Wt[n][k]
// ---------------------------------------------------------------------------
__global__ void k_conv_wtT(const float* __restrict__ W,
                           unsigned short* __restrict__ th_out,
                           unsigned short* __restrict__ tl_out) {
    __shared__ unsigned short th[32][33], tl[32][33];
    int k0 = blockIdx.y * 32, n0 = blockIdx.x * 32;
    int tx = threadIdx.x, ty = threadIdx.y;        // 32 x 8
#pragma unroll
    for (int i = 0; i < 4; ++i) {
        int k = ty + i * 8;
        float f = W[(size_t)(k0 + k) * 1024 + n0 + tx];
        unsigned short h, l;
        bf16_split(f, h, l);
        th[tx][k] = h; tl[tx][k] = l;              // store transposed: [n][k]
    }
    __syncthreads();
#pragma unroll
    for (int i = 0; i < 4; ++i) {
        int n = ty + i * 8;
        th_out[(size_t)(n0 + n) * 1024 + k0 + tx] = th[n][tx];
        tl_out[(size_t)(n0 + n) * 1024 + k0 + tx] = tl[n][tx];
    }
}

// ---------------------------------------------------------------------------
// WMMA GEMM: C[2048,1024] = A * W, fp32 via bf16x3 planes.
// A planes: [2048,1024] row-major. B planes: Wt[n][k] (pre-transposed).
// 128x128 tile, BK=32, 8 waves (2x4), wave tile 64x32.
// Tiles staged with global_load_async_to_lds_b128, double-buffered.
// mode: 0 = plain store, 1 = silu store.
// ---------------------------------------------------------------------------
__global__ __launch_bounds__(256) void k_gemm_async(
    const unsigned short* __restrict__ Ah_g, const unsigned short* __restrict__ Al_g,
    const unsigned short* __restrict__ Bth_g, const unsigned short* __restrict__ Btl_g,
    float* __restrict__ Cout, int mode) {

    // [buf][tile: Ahi,Alo,Bthi,Btlo][128 rows x 32 k]  = 64 KB
    __shared__ __align__(16) unsigned short tiles[2][4][128 * 32];

    const int tid  = threadIdx.x;
    const int lane = tid & 31;
    const int wid  = tid >> 5;
    const int half = lane >> 4;
    const int lrow = lane & 15;
    const int wm   = wid & 1;
    const int wn   = wid >> 1;
    const int gM   = blockIdx.y * 128;
    const int gN   = blockIdx.x * 128;

    const unsigned short* srcs[4] = { Ah_g, Al_g, Bth_g, Btl_g };
    const int rbase[4] = { gM, gM, gN, gN };

    v8f acc[4][2];
#pragma unroll
    for (int i = 0; i < 4; ++i)
#pragma unroll
        for (int j = 0; j < 2; ++j)
#pragma unroll
            for (int e = 0; e < 8; ++e) acc[i][j][e] = 0.f;

    // Stage K-step `ks` into LDS buffer `buf`: 4 tiles x 2 x 16B chunks/thread.
    auto stage = [&](int buf, int ks) {
        int k0 = ks * 32;
#pragma unroll
        for (int tbl = 0; tbl < 4; ++tbl) {
#pragma unroll
            for (int it = 0; it < 2; ++it) {
                int cidx = tid + it * 256;               // 0..511
                int row  = cidx >> 2;                    // 4 chunks per row
                int seg  = (cidx & 3) << 3;              // element offset, 16B chunks
                const void* g = (const void*)(srcs[tbl] +
                                (size_t)(rbase[tbl] + row) * 1024 + k0 + seg);
                unsigned loff = (unsigned)(uintptr_t)(const void*)
                                &tiles[buf][tbl][row * 32 + seg];
                asm volatile("global_load_async_to_lds_b128 %0, %1, off"
                             :: "v"(loff), "v"(g) : "memory");
            }
        }
    };

    stage(0, 0);

    for (int ks = 0; ks < 32; ++ks) {
        const int cur = ks & 1;
        if (ks + 1 < 32) {
            stage(cur ^ 1, ks + 1);                      // prefetch next K-step
            asm volatile("s_wait_asynccnt 0x8" ::: "memory");  // cur buffer landed
        } else {
            asm volatile("s_wait_asynccnt 0x0" ::: "memory");
        }
        __syncthreads();

        const unsigned short* Ahi = tiles[cur][0];
        const unsigned short* Alo = tiles[cur][1];
        const unsigned short* Bhi = tiles[cur][2];
        const unsigned short* Blo = tiles[cur][3];

        // B fragments (lanes0-15: K=0..15 | lanes16-31: K=16..31, contiguous)
        Frag bh[2], bl[2];
#pragma unroll
        for (int ns = 0; ns < 2; ++ns) {
            int n = wn * 32 + ns * 16 + lrow;
            int kb = half * 16;
            bh[ns].u4[0] = *(const uint4*)&Bhi[n * 32 + kb];
            bh[ns].u4[1] = *(const uint4*)&Bhi[n * 32 + kb + 8];
            bl[ns].u4[0] = *(const uint4*)&Blo[n * 32 + kb];
            bl[ns].u4[1] = *(const uint4*)&Blo[n * 32 + kb + 8];
        }
        // A fragments (lanes0-15: K=0..15 | lanes16-31: K=16..31, contiguous)
#pragma unroll
        for (int ms = 0; ms < 4; ++ms) {
            int mr = wm * 64 + ms * 16 + lrow;
            Frag ah, al;
            ah.u4[0] = *(const uint4*)&Ahi[mr * 32 + half * 16];
            ah.u4[1] = *(const uint4*)&Ahi[mr * 32 + half * 16 + 8];
            al.u4[0] = *(const uint4*)&Alo[mr * 32 + half * 16];
            al.u4[1] = *(const uint4*)&Alo[mr * 32 + half * 16 + 8];
#pragma unroll
            for (int ns = 0; ns < 2; ++ns) {
                acc[ms][ns] = __builtin_amdgcn_wmma_f32_16x16x32_bf16(
                    false, ah.v, false, bh[ns].v, (short)0, acc[ms][ns], false, false);
                acc[ms][ns] = __builtin_amdgcn_wmma_f32_16x16x32_bf16(
                    false, ah.v, false, bl[ns].v, (short)0, acc[ms][ns], false, false);
                acc[ms][ns] = __builtin_amdgcn_wmma_f32_16x16x32_bf16(
                    false, al.v, false, bh[ns].v, (short)0, acc[ms][ns], false, false);
            }
        }
        __syncthreads();   // all waves done reading `cur` before it is re-filled
    }

    // Epilogue: C layout => VGPR r, lane -> (M = r + 8*half, N = lrow)
#pragma unroll
    for (int ms = 0; ms < 4; ++ms) {
#pragma unroll
        for (int ns = 0; ns < 2; ++ns) {
            int row0 = gM + wm * 64 + ms * 16 + half * 8;
            int col  = gN + wn * 32 + ns * 16 + lrow;
#pragma unroll
            for (int r = 0; r < 8; ++r) {
                float val = acc[ms][ns][r];
                if (mode == 1) val = val / (1.f + expf(-val));   // SiLU
                Cout[(size_t)(row0 + r) * 1024 + col] = val;
            }
        }
    }
}

// ---------------------------------------------------------------------------
// WKV scan: one block per (b,h), 64 threads (thread j = v index).
// ---------------------------------------------------------------------------
__global__ __launch_bounds__(64) void k_scan(const float* __restrict__ r,
                                             const float* __restrict__ k,
                                             const float* __restrict__ v,
                                             const float* __restrict__ w,
                                             const float* __restrict__ u,
                                             float* __restrict__ y) {
    int bh = blockIdx.x;            // 0..31
    int b = bh >> 4, h = bh & 15;
    int j = threadIdx.x;            // 0..63
    __shared__ float sr[64], sk[64], sw[64], su[64];
    su[j] = u[h * 64 + j];
    float state[64];
#pragma unroll
    for (int i = 0; i < 64; ++i) state[i] = 0.f;
    const size_t base = (size_t)(b * 1024) * 1024 + h * 64;
    for (int t = 0; t < 1024; ++t) {
        size_t off = base + (size_t)t * 1024;
        __syncthreads();
        sr[j] = r[off + j]; sk[j] = k[off + j]; sw[j] = w[off + j];
        __syncthreads();
        float vj = v[off + j];
        float yj = 0.f;
#pragma unroll
        for (int i = 0; i < 64; ++i) {
            float kv = sk[i] * vj;
            yj += sr[i] * (state[i] + su[i] * kv);
            state[i] = state[i] * sw[i] + kv;
        }
        y[off + j] = yj;
    }
}

// ---------------------------------------------------------------------------
// GroupNorm per head (K=64) + gate by g (already SiLU'd):  yg = GN(y)*g
// ---------------------------------------------------------------------------
__global__ __launch_bounds__(1024) void k_gn_gate(const float* __restrict__ y,
                                                  const float* __restrict__ g,
                                                  const float* __restrict__ lnw,
                                                  const float* __restrict__ lnb,
                                                  float* __restrict__ yg) {
    int row = blockIdx.x;           // 0..2047
    int c = threadIdx.x;            // 0..1023
    __shared__ float sy[1024];
    __shared__ float smu[16], srs[16];
    size_t idx = (size_t)row * 1024 + c;
    float val = y[idx];
    sy[c] = val;
    __syncthreads();
    if (c < 16) {
        float s = 0.f, s2 = 0.f;
        for (int i = 0; i < 64; ++i) { float t = sy[c * 64 + i]; s += t; s2 += t * t; }
        float mu = s * (1.f / 64.f);
        float var = s2 * (1.f / 64.f) - mu * mu;
        smu[c] = mu;
        srs[c] = rsqrtf(var + 1e-5f);
    }
    __syncthreads();
    int h = c >> 6;
    float yn = (val - smu[h]) * srs[h] * lnw[c] + lnb[c];
    yg[idx] = yn * g[idx];
}

// ---------------------------------------------------------------------------
extern "C" void kernel_launch(void* const* d_in, const int* in_sizes, int n_in,
                              void* d_out, int out_size, void* d_ws, size_t ws_size,
                              hipStream_t stream) {
    (void)in_sizes; (void)n_in; (void)out_size; (void)ws_size;

    const float* x      = (const float*)d_in[0];
    const float* tmx    = (const float*)d_in[1];
    const float* mu5    = (const float*)d_in[2];
    const float* tmw1   = (const float*)d_in[3];
    const float* tmw2   = (const float*)d_in[4];
    const float* tdec   = (const float*)d_in[5];
    const float* tdw1   = (const float*)d_in[6];
    const float* tdw2   = (const float*)d_in[7];
    const float* u      = (const float*)d_in[8];
    const float* Wr     = (const float*)d_in[9];
    const float* Wk     = (const float*)d_in[10];
    const float* Wv     = (const float*)d_in[11];
    const float* Wg     = (const float*)d_in[12];
    const float* Wo     = (const float*)d_in[13];
    const float* lnw    = (const float*)d_in[14];
    const float* lnb    = (const float*)d_in[15];
    float* out          = (float*)d_out;

    const size_t MC = 2048ull * 1024ull;    // 2,097,152
    float* ws  = (float*)d_ws;
    float* SX   = ws;                        // [M,C]   (reused later as YG)
    float* XXX  = ws + MC;                   // [M,C]   (reused later as Y)
    float* M160 = ws + 2 * MC;               // [M,160]
    float* XW   = M160 + 2048 * 160;         // 5 x [M,C]
    float* XK   = XW + MC;
    float* XV   = XW + 2 * MC;
    float* XR   = XW + 3 * MC;
    float* XG   = XW + 4 * MC;
    float* T1   = XW + 5 * MC;               // [M,64]
    float* WD   = T1 + 2048 * 64;            // [M,C]  decay w
    float* Rb   = WD + MC;                   // [M,C]
    float* Kb   = Rb + MC;
    float* Vb   = Kb + MC;
    float* Gb   = Vb + MC;
    float* Yb   = XXX;                       // reuse
    float* YG   = SX;                        // reuse
    // bf16 plane scratch (reused per projection)
    unsigned short* APH = (unsigned short*)(Gb + MC);        // [2048,1024] bf16
    unsigned short* APL = APH + MC;                          // [2048,1024] bf16
    unsigned short* WTH = APL + MC;                          // [1024,1024] bf16 (transposed)
    unsigned short* WTL = WTH + 1024 * 1024;                 // [1024,1024] bf16

    // 1. token shift
    k_mix1<<<8192, 256, 0, stream>>>(x, tmx, SX, XXX);
    // 2. m = tanh(xxx @ tm_w1)
    k_lora1<<<2048, 160, 0, stream>>>(XXX, tmw1, M160);
    // 3. 5-way mix
    k_mix2<<<dim3(4, 2048), 256, 0, stream>>>(x, SX, M160, mu5, tmw2,
                                              XW, XK, XV, XR, XG);
    // 4-5. decay LoRA -> w
    k_decay1<<<2048, 64, 0, stream>>>(XW, tdw1, T1);
    k_decay2<<<dim3(4, 2048), 256, 0, stream>>>(T1, tdw2, tdec, WD);

    // 6. big projections via async-staged WMMA GEMMs
    dim3 ggrid(8, 16);
    dim3 tgrid(32, 32), tblk(32, 8);
    const float* Ain[4]  = { XR, XK, XV, XG };
    const float* Win[4]  = { Wr, Wk, Wv, Wg };
    float*       Cout[4] = { Rb, Kb, Vb, Gb };
    const int    emode[4] = { 0, 0, 0, 1 };                  // SiLU on g
    for (int p = 0; p < 4; ++p) {
        k_conv_act<<<4096, 256, 0, stream>>>(Ain[p], APH, APL);
        k_conv_wtT<<<tgrid, tblk, 0, stream>>>(Win[p], WTH, WTL);
        k_gemm_async<<<ggrid, 256, 0, stream>>>(APH, APL, WTH, WTL, Cout[p], emode[p]);
    }

    // 7. WKV recurrence
    k_scan<<<32, 64, 0, stream>>>(Rb, Kb, Vb, WD, u, Yb);

    // 8. GroupNorm + gate
    k_gn_gate<<<2048, 1024, 0, stream>>>(Yb, Gb, lnw, lnb, YG);

    // 9. output projection -> d_out
    k_conv_act<<<4096, 256, 0, stream>>>(YG, APH, APL);
    k_conv_wtT<<<tgrid, tblk, 0, stream>>>(Wo, WTH, WTL);
    k_gemm_async<<<ggrid, 256, 0, stream>>>(APH, APL, WTH, WTL, out, 0);
}
